// MultiHeadAttention_76854144795234
// MI455X (gfx1250) — matrix operand order repaired and checked by reading, not compiled, not run on hardware
//
#include <hip/hip_runtime.h>
#include <hip/hip_bf16.h>

// ---------------- CDNA5 WMMA plumbing (gfx1250, wave32) ----------------
typedef __attribute__((ext_vector_type(16))) __bf16 bf16x16;
typedef __attribute__((ext_vector_type(8)))  __bf16 bf16x8;
typedef __attribute__((ext_vector_type(8)))  float  f32x8;
typedef __attribute__((ext_vector_type(4)))  float  f32x4;
typedef __attribute__((ext_vector_type(4)))  unsigned short u16x4;

union FragU { bf16x16 v; bf16x8 h[2]; };

__device__ __forceinline__ unsigned short f2u16(float f) {
  return __builtin_bit_cast(unsigned short, (__bf16)f);   // HW cvt, RNE
}
__device__ __forceinline__ float bf16u_to_f32(unsigned short h) {
  unsigned u = ((unsigned)h) << 16;
  return __builtin_bit_cast(float, u);
}
__device__ __forceinline__ f32x8 wmma_bf16(bf16x16 a, bf16x16 b, f32x8 c) {
  return __builtin_amdgcn_wmma_f32_16x16x32_bf16(false, a, false, b, (short)0, c,
                                                 false, false);
}
// Fragment maps per ISA 7.12.2 (wave32):
//  A 16x32: lane%16 = row; lanes<16 K{0..7,16..23}, lanes>=16 K{8..15,24..31}
__device__ __forceinline__ bf16x16 load_a_bf16(const unsigned short* p, int hi) {
  FragU u;
  u.h[0] = *(const bf16x8*)(p + 8 * hi);
  u.h[1] = *(const bf16x8*)(p + 8 * hi + 16);
  return u.v;
}
//  B 32x16: lane%16 = col; lanes<16 K{0..7,16..23}, lanes>=16 K{8..15,24..31}
__device__ __forceinline__ bf16x16 load_b_bf16(const unsigned short* p, int hi) {
  FragU u;
  u.h[0] = *(const bf16x8*)(p + 8 * hi);
  u.h[1] = *(const bf16x8*)(p + 8 * hi + 16);
  return u.v;
}
__device__ __forceinline__ bf16x16 load_a_f32(const float* p, int hi) {
  f32x8 lo = *(const f32x8*)(p + 8 * hi);
  f32x8 hh = *(const f32x8*)(p + 8 * hi + 16);
  FragU u;
#pragma unroll
  for (int e = 0; e < 8; ++e) { u.h[0][e] = (__bf16)lo[e]; u.h[1][e] = (__bf16)hh[e]; }
  return u.v;
}
//  C/D 16x16 f32: lane%16 = col N; VGPR r -> row M = r + 8*(lane>=16)

#define BB    4
#define SS    1024
#define EMB   1024
#define HH    16
#define HD    64
#define BH    64
#define MAXP  512
#define NREL  1025
#define PSTR  1040          // P/R row stride (elements)
#define RVSTR 1056          // padded rel_v^T row stride

// =====================================================================
// Setup kernels: one-time bf16 conversion / transposition into ws
// =====================================================================
__global__ __launch_bounds__(256) void cvt_bf16_kernel(
    const float* __restrict__ src, unsigned short* __restrict__ dst, int n4) {
  int i = blockIdx.x * 256 + threadIdx.x;
  if (i >= n4) return;
  f32x4 f = ((const f32x4*)src)[i];
  u16x4 o;
#pragma unroll
  for (int e = 0; e < 4; ++e) o[e] = f2u16(f[e]);
  ((u16x4*)dst)[i] = o;
}

// W[1024,1024] f32 -> WT[1024,1024] bf16 (WT[n][k] = W[k][n]); z selects matrix
__global__ __launch_bounds__(256) void transpose_w_kernel(
    const float* __restrict__ W0, const float* __restrict__ W1,
    const float* __restrict__ W2, const float* __restrict__ W3,
    unsigned short* __restrict__ T0, unsigned short* __restrict__ T1,
    unsigned short* __restrict__ T2, unsigned short* __restrict__ T3) {
  const float* W; unsigned short* T;
  if (blockIdx.z == 0)      { W = W0; T = T0; }
  else if (blockIdx.z == 1) { W = W1; T = T1; }
  else if (blockIdx.z == 2) { W = W2; T = T2; }
  else                      { W = W3; T = T3; }
  __shared__ unsigned short tile[32][33];
  const int n0 = blockIdx.x * 32, k0 = blockIdx.y * 32;
  const int tx = threadIdx.x & 31, ty = threadIdx.x >> 5;   // 8 rows / iter
  for (int r = ty; r < 32; r += 8)
    tile[r][tx] = f2u16(W[(size_t)(k0 + r) * EMB + n0 + tx]);
  __syncthreads();
  for (int r = ty; r < 32; r += 8)
    T[(size_t)(n0 + r) * EMB + k0 + tx] = tile[tx][r];
}

// rel_v[1025,64] f32 -> rvt[64,1056] bf16, zero-padded beyond j=1024
__global__ __launch_bounds__(256) void transpose_relv_kernel(
    const float* __restrict__ rel_v, unsigned short* __restrict__ rvt) {
  int idx = blockIdx.x * 256 + threadIdx.x;
  if (idx >= HD * RVSTR) return;
  int d = idx / RVSTR, j = idx % RVSTR;
  float v = (j < NREL) ? rel_v[(size_t)j * HD + d] : 0.f;
  rvt[idx] = f2u16(v);
}

// vws[BH*S,64] bf16 -> vt[BH*64,S] bf16
__global__ __launch_bounds__(256) void transpose_v_kernel(
    const unsigned short* __restrict__ v, unsigned short* __restrict__ vt) {
  __shared__ unsigned short tile[64][65];
  const int bh = blockIdx.y, s0 = blockIdx.x * 64;
  const int tx = threadIdx.x & 63, ty = threadIdx.x >> 6;   // 4 rows / iter
  for (int r = ty; r < 64; r += 4)
    tile[r][tx] = v[((size_t)bh * SS + s0 + r) * HD + tx];
  __syncthreads();
  for (int r = ty; r < 64; r += 4)
    vt[((size_t)bh * HD + r) * SS + s0 + tx] = tile[tx][r];
}

// =====================================================================
// Kernel: QKV projections, all-bf16, no LDS (WT chunks live in WGP$/L2).
// grid (N/64, M/256, 3), 256 thr = 8 waves, wave -> 32x64 tile
// (two 16-row A fragments reuse each B fragment -> 1.5 loads per WMMA).
// =====================================================================
__global__ __launch_bounds__(256) void qkv_gemm_kernel(
    const unsigned short* __restrict__ xq, const unsigned short* __restrict__ xk,
    const unsigned short* __restrict__ xv,
    const unsigned short* __restrict__ WqT, const unsigned short* __restrict__ WkT,
    const unsigned short* __restrict__ WvT,
    const float* __restrict__ bq, const float* __restrict__ bk,
    const float* __restrict__ bv,
    unsigned short* __restrict__ qws, unsigned short* __restrict__ kws,
    unsigned short* __restrict__ vws) {
  const int tid = threadIdx.x, lane = tid & 31, wv = tid >> 5;
  const int hi = (lane >= 16) ? 1 : 0, ln = lane & 15;
  const int nBase = blockIdx.x * 64;
  const int mBase = blockIdx.y * 256 + wv * 32;

  const unsigned short* X; const unsigned short* WT; const float* bias;
  unsigned short* dst;
  if (blockIdx.z == 0)      { X = xq; WT = WqT; bias = bq; dst = qws; }
  else if (blockIdx.z == 1) { X = xk; WT = WkT; bias = bk; dst = kws; }
  else                      { X = xv; WT = WvT; bias = bv; dst = vws; }

  const unsigned short* arow0 = X + (size_t)(mBase + ln) * EMB;
  const unsigned short* arow1 = X + (size_t)(mBase + 16 + ln) * EMB;
  f32x8 acc0[4] = {}, acc1[4] = {};

  for (int k0 = 0; k0 < EMB; k0 += 32) {
    bf16x16 a0 = load_a_bf16(arow0 + k0, hi);
    bf16x16 a1 = load_a_bf16(arow1 + k0, hi);
#pragma unroll
    for (int s = 0; s < 4; ++s) {
      const int n = nBase + s * 16 + ln;
      bf16x16 b = load_b_bf16(WT + (size_t)n * EMB + k0, hi);
      acc0[s] = wmma_bf16(a0, b, acc0[s]);
      acc1[s] = wmma_bf16(a1, b, acc1[s]);
    }
  }
#pragma unroll
  for (int s = 0; s < 4; ++s)
#pragma unroll
    for (int r = 0; r < 8; ++r) {
      int n = nBase + s * 16 + ln;
      int h = n >> 6, d = n & 63;
      {
        int m = mBase + r + hi * 8;
        float y = acc0[s][r] + bias[n];
        int b = m >> 10, sp = m & 1023;
        dst[(((size_t)(b * HH + h)) * SS + sp) * HD + d] = f2u16(y);
      }
      {
        int m = mBase + 16 + r + hi * 8;
        float y = acc1[s][r] + bias[n];
        int b = m >> 10, sp = m & 1023;
        dst[(((size_t)(b * HH + h)) * SS + sp) * HD + d] = f2u16(y);
      }
    }
}

// =====================================================================
// Attention per (16-row q-block, head). grid (S/16, BH), 128 thr = 4 waves.
// =====================================================================
__global__ __launch_bounds__(128) void attn_kernel(
    const unsigned short* __restrict__ qws, const unsigned short* __restrict__ kws,
    const unsigned short* __restrict__ vt,
    const unsigned short* __restrict__ relkb, const unsigned short* __restrict__ rvt,
    unsigned short* __restrict__ ctx, float* __restrict__ attn_g) {
  const int tid = threadIdx.x, lane = tid & 31, wv = tid >> 5;
  const int hi = (lane >= 16) ? 1 : 0, ln = lane & 15;
  const int q0 = blockIdx.x * 16;
  const int bh = blockIdx.y;

  __shared__ unsigned short lds_p[16 * PSTR + 32];  // R (bf16), later P (bf16)
  __shared__ float red_m[4][16], red_s[4][16];
  __shared__ float row_m[16], row_is[16];
  __shared__ float edge_lo[16], edge_hi[16];

  const unsigned short* qrow = qws + (size_t)(bh * SS + q0 + ln) * HD;
  bf16x16 aq0 = load_a_bf16(qrow, hi);
  bf16x16 aq1 = load_a_bf16(qrow + 32, hi);

  // ---- Phase R: R[m,j] = q_m . rel_k[j]  (65 j-tiles of 16) ----
  for (int jt = wv; jt < 65; jt += 4) {
    int j0 = jt * 16;
    int jcol = j0 + ln;
    int jc = jcol > (NREL - 1) ? (NREL - 1) : jcol;  // j>1024 never gathered
    const unsigned short* rk = relkb + (size_t)jc * HD;
    f32x8 c = {};
    c = wmma_bf16(aq0, load_b_bf16(rk, hi), c);
    c = wmma_bf16(aq1, load_b_bf16(rk + 32, hi), c);
#pragma unroll
    for (int r = 0; r < 8; ++r)
      lds_p[(r + hi * 8) * PSTR + j0 + ln] = f2u16(c[r]);
  }
  __syncthreads();

  // ---- Pass 1: logits = (q.k + R[clip])/8, online row max/sum ----
  float mx[8], sm[8];
#pragma unroll
  for (int r = 0; r < 8; ++r) { mx[r] = -3.0e38f; sm[r] = 0.f; }

  for (int kt = wv; kt < 64; kt += 4) {
    int k0 = kt * 16;
    const unsigned short* krow = kws + (size_t)(bh * SS + k0 + ln) * HD;
    f32x8 c = {};
    c = wmma_bf16(aq0, load_b_bf16(krow, hi), c);
    c = wmma_bf16(aq1, load_b_bf16(krow + 32, hi), c);

    int kcol = k0 + ln;
#pragma unroll
    for (int r = 0; r < 8; ++r) {
      int m = r + hi * 8;
      int dd = kcol - (q0 + m);
      dd = dd < -MAXP ? -MAXP : (dd > MAXP ? MAXP : dd);
      float R = bf16u_to_f32(lds_p[m * PSTR + dd + MAXP]);
      float l = (c[r] + R) * 0.125f;
      attn_g[(size_t)(bh * SS + q0 + m) * SS + kcol] = l;   // stash raw logit
      float nm = fmaxf(mx[r], l);
      sm[r] = sm[r] * __expf(mx[r] - nm) + __expf(l - nm);
      mx[r] = nm;
    }
  }
#pragma unroll
  for (int off = 1; off < 16; off <<= 1)
#pragma unroll
    for (int r = 0; r < 8; ++r) {
      float om = __shfl_xor(mx[r], off);
      float os = __shfl_xor(sm[r], off);
      float nm = fmaxf(mx[r], om);
      sm[r] = sm[r] * __expf(mx[r] - nm) + os * __expf(om - nm);
      mx[r] = nm;
    }
  if (ln == 0)
#pragma unroll
    for (int r = 0; r < 8; ++r) { red_m[wv][r + hi * 8] = mx[r]; red_s[wv][r + hi * 8] = sm[r]; }
  __syncthreads();

  if (tid < 16) {
    float m = red_m[0][tid], s = red_s[0][tid];
#pragma unroll
    for (int w = 1; w < 4; ++w) {
      float om = red_m[w][tid], os = red_s[w][tid];
      float nm = fmaxf(m, om);
      s = s * __expf(m - nm) + os * __expf(om - nm);
      m = nm;
    }
    row_m[tid] = m;
    row_is[tid] = 1.0f / s;
    edge_lo[tid] = 0.f; edge_hi[tid] = 0.f;
  }
  for (int idx = tid; idx < 16 * PSTR + 32; idx += 128) lds_p[idx] = 0;
  __threadfence();
  __syncthreads();

  // ---- Pass 2: normalize attn in place, scatter into P ----
  for (int idx = tid; idx < 16 * SS; idx += 128) {
    int m = idx >> 10, k = idx & (SS - 1);
    size_t gi = (size_t)(bh * SS + q0 + m) * SS + k;
    float a = __expf(attn_g[gi] - row_m[m]) * row_is[m];
    attn_g[gi] = a;
    int dd = k - (q0 + m);
    if (dd <= -MAXP)      atomicAdd(&edge_lo[m], a);
    else if (dd >= MAXP)  atomicAdd(&edge_hi[m], a);
    else                  lds_p[m * PSTR + dd + MAXP] = f2u16(a);
  }
  __threadfence();
  __syncthreads();
  if (tid < 16) {
    lds_p[tid * PSTR + 0]    = f2u16(edge_lo[tid]);
    lds_p[tid * PSTR + 1024] = f2u16(edge_hi[tid]);
  }
  __syncthreads();

  // ---- ctx = attn @ v  +  P @ rel_v   (wave wv owns d-subtile) ----
  const int d0 = wv * 16;
  f32x8 c = {};
  const float* arow = attn_g + (size_t)(bh * SS + q0 + ln) * SS;
  const unsigned short* vrow = vt + (size_t)(bh * HD + d0 + ln) * SS;
  for (int kc = 0; kc < 32; ++kc) {
    int k0 = kc * 32;
    bf16x16 a = load_a_f32(arow + k0, hi);
    bf16x16 b = load_b_bf16(vrow + k0, hi);
    c = wmma_bf16(a, b, c);
  }
  const unsigned short* rvrow = rvt + (size_t)(d0 + ln) * RVSTR;
  const unsigned short* prow  = lds_p + ln * PSTR;
  for (int jc = 0; jc < 33; ++jc) {
    int j0 = jc * 32;
    bf16x16 a = load_a_bf16(prow + j0, hi);
    bf16x16 b = load_b_bf16(rvrow + j0, hi);
    c = wmma_bf16(a, b, c);
  }

  // scrambled head-merge faithful to reference reshape(H,B,...) quirk
  const int i0 = bh >> 2, i1 = bh & 3;
#pragma unroll
  for (int r = 0; r < 8; ++r) {
    int sp = q0 + r + hi * 8;
    size_t G = (size_t)i0 * SS + sp;
    ctx[G * 256 + i1 * 64 + d0 + ln] = f2u16(c[r]);
  }
}

// =====================================================================
// Kernel: out = ctx(bf16) @ Wo^T(bf16) + bo -> f32. grid (N/64, M/256)
// =====================================================================
__global__ __launch_bounds__(256) void out_gemm_kernel(
    const unsigned short* __restrict__ ctx, const unsigned short* __restrict__ WoT,
    const float* __restrict__ bo, float* __restrict__ out) {
  const int tid = threadIdx.x, lane = tid & 31, wv = tid >> 5;
  const int hi = (lane >= 16) ? 1 : 0, ln = lane & 15;
  const int nBase = blockIdx.x * 64;
  const int mBase = blockIdx.y * 256 + wv * 32;

  const unsigned short* arow0 = ctx + (size_t)(mBase + ln) * EMB;
  const unsigned short* arow1 = ctx + (size_t)(mBase + 16 + ln) * EMB;
  f32x8 acc0[4] = {}, acc1[4] = {};

  for (int k0 = 0; k0 < EMB; k0 += 32) {
    bf16x16 a0 = load_a_bf16(arow0 + k0, hi);
    bf16x16 a1 = load_a_bf16(arow1 + k0, hi);
#pragma unroll
    for (int s = 0; s < 4; ++s) {
      const int n = nBase + s * 16 + ln;
      bf16x16 b = load_b_bf16(WoT + (size_t)n * EMB + k0, hi);
      acc0[s] = wmma_bf16(a0, b, acc0[s]);
      acc1[s] = wmma_bf16(a1, b, acc1[s]);
    }
  }
#pragma unroll
  for (int s = 0; s < 4; ++s)
#pragma unroll
    for (int r = 0; r < 8; ++r) {
      int n = nBase + s * 16 + ln;
      out[(size_t)(mBase + r + hi * 8) * EMB + n]      = acc0[s][r] + bo[n];
      out[(size_t)(mBase + 16 + r + hi * 8) * EMB + n] = acc1[s][r] + bo[n];
    }
}

// =====================================================================
extern "C" void kernel_launch(void* const* d_in, const int* in_sizes, int n_in,
                              void* d_out, int out_size, void* d_ws, size_t ws_size,
                              hipStream_t stream) {
  (void)in_sizes; (void)n_in; (void)out_size; (void)ws_size;
  const float* query = (const float*)d_in[0];
  const float* key_  = (const float*)d_in[1];
  const float* value = (const float*)d_in[2];
  const float* Wq = (const float*)d_in[3];  const float* bq = (const float*)d_in[4];
  const float* Wk = (const float*)d_in[5];  const float* bk = (const float*)d_in[6];
  const float* Wv = (const float*)d_in[7];  const float* bv = (const float*)d_in[8];
  const float* Wo = (const float*)d_in[9];  const float* bo = (const float*)d_in[10];
  const float* rel_k = (const float*)d_in[11];
  const float* rel_v = (const float*)d_in[12];

  float* out  = (float*)d_out;
  float* attn = out + (size_t)BB * SS * EMB;

  const size_t M4 = (size_t)4 * 1024 * 1024;   // elements in a 4096x1024 matrix
  const size_t M1 = (size_t)1024 * 1024;
  unsigned short* p = (unsigned short*)d_ws;
  unsigned short* xq    = p;  p += M4;
  unsigned short* xk    = p;  p += M4;
  unsigned short* xv    = p;  p += M4;
  unsigned short* WqT   = p;  p += M1;
  unsigned short* WkT   = p;  p += M1;
  unsigned short* WvT   = p;  p += M1;
  unsigned short* WoT   = p;  p += M1;
  unsigned short* relkb = p;  p += (size_t)NREL * HD;        // 65600 (16-mult)
  unsigned short* rvt   = p;  p += (size_t)HD * RVSTR;       // 67584
  unsigned short* qws   = p;  p += M4;
  unsigned short* kws   = p;  p += M4;
  unsigned short* vws   = p;  p += M4;
  unsigned short* vt    = p;  p += M4;
  unsigned short* ctx   = p;  p += M4;

  // -- setup: bf16 conversions & transposes --
  cvt_bf16_kernel<<<dim3((M4 / 4 + 255) / 256), 256, 0, stream>>>(query, xq, (int)(M4 / 4));
  cvt_bf16_kernel<<<dim3((M4 / 4 + 255) / 256), 256, 0, stream>>>(key_, xk, (int)(M4 / 4));
  cvt_bf16_kernel<<<dim3((M4 / 4 + 255) / 256), 256, 0, stream>>>(value, xv, (int)(M4 / 4));
  cvt_bf16_kernel<<<dim3((NREL * HD / 4 + 255) / 256), 256, 0, stream>>>(rel_k, relkb, NREL * HD / 4);
  transpose_w_kernel<<<dim3(32, 32, 4), 256, 0, stream>>>(Wq, Wk, Wv, Wo, WqT, WkT, WvT, WoT);
  transpose_relv_kernel<<<dim3((HD * RVSTR + 255) / 256), 256, 0, stream>>>(rel_v, rvt);

  // -- pipeline --
  qkv_gemm_kernel<<<dim3(EMB / 64, (BB * SS) / 256, 3), 256, 0, stream>>>(
      xq, xk, xv, WqT, WkT, WvT, bq, bk, bv, qws, kws, vws);
  transpose_v_kernel<<<dim3(SS / 64, BH), 256, 0, stream>>>(vws, vt);
  attn_kernel<<<dim3(SS / 16, BH), 128, 0, stream>>>(
      qws, kws, vt, relkb, rvt, ctx, attn);
  out_gemm_kernel<<<dim3(EMB / 64, (BB * SS) / 256), 256, 0, stream>>>(
      ctx, WoT, bo, out);
}
